// Seq2Seq_57990648431226
// MI455X (gfx1250) — matrix operand order repaired and checked by reading, not compiled, not run on hardware
//
#include <hip/hip_runtime.h>
#include <stddef.h>

// ---------------- types ----------------
typedef __attribute__((ext_vector_type(16))) __bf16 v16bf;
typedef __attribute__((ext_vector_type(8)))  __bf16 v8bf;
typedef __attribute__((ext_vector_type(8)))  float  v8f;

union ABu { v16bf v; v8bf h[2]; };

#define WMMA_BF16(A, B, C) \
  __builtin_amdgcn_wmma_f32_16x16x32_bf16(false, (A), false, (B), (short)0, (C), false, false)

// Scheduling fence: stops LLVM from hoisting loop-invariant weight loads out of
// the step loops (which previously caused massive VGPR spills to scratch).
#define SCHED_FENCE() asm volatile("" ::: "memory")

__device__ __forceinline__ float sigmf(float x) { return 1.0f / (1.0f + __expf(-x)); }

// ---------------- fp32 -> bf16 convert ----------------
__global__ __launch_bounds__(256) void cvt_bf16_kernel(const float* __restrict__ in,
                                                       __bf16* __restrict__ out, size_t n) {
  size_t i = (size_t)blockIdx.x * 256 + threadIdx.x;
  if (i < n) out[i] = (__bf16)in[i];
}

// ---------------- embedding gather: x[t][b][e] = emb[tok[b][t]][e], as bf16 -------------
__global__ __launch_bounds__(256) void embed_kernel(const int* __restrict__ tok, int tok_stride,
                                                    const float* __restrict__ emb,
                                                    __bf16* __restrict__ xout, int L) {
  size_t tid = (size_t)blockIdx.x * 256 + threadIdx.x;
  size_t total = (size_t)L * 32 * 256;
  if (tid >= total) return;
  int e = (int)(tid & 255);
  int b = (int)((tid >> 8) & 31);
  int t = (int)(tid >> 13);
  int token = tok[b * tok_stride + t];
  xout[tid] = (__bf16)emb[(size_t)token * 256 + e];
}

// ---------------- persistent single-workgroup LSTM scan ----------------
// x_seq: [L][32][256] bf16;  Wih/Whh: [1024][256] bf16 (row n = gate output n)
// gates n = g*256 + u, PyTorch order i,f,g,o.
// One block of 1024 threads = 32 wave32s. Wave w owns batch tile m = w&1 and
// unit tile u16 = w>>1: 4 gate accumulators + 1 c tile per wave.
// Weights stream from L2 every step (1 MB, L2-resident); SCHED_FENCE keeps the
// compiler from hoisting them into (spilled) registers.
__global__ __launch_bounds__(1024) void lstm_kernel(const __bf16* __restrict__ x_seq,
                                                    const __bf16* __restrict__ Wih,
                                                    const __bf16* __restrict__ Whh,
                                                    const float* __restrict__ bias,
                                                    const float* __restrict__ hc_in, int init_zero,
                                                    float* __restrict__ hc_out,
                                                    __bf16* __restrict__ hs_out, int L) {
  __shared__ __bf16 h_sh[32 * 256];  // 16 KB
  const int tid = threadIdx.x;
  const int w = tid >> 5, lane = tid & 31;
  const int half = lane >> 4, lr = lane & 15;
  const int m = w & 1;       // batch tile (rows m*16 .. m*16+15)
  const int u16 = w >> 1;    // unit tile (units u16*16 .. +15)
  const int u = u16 * 16 + lr;

  // loop-invariant per-lane bias values (one per gate)
  float bv[4];
#pragma unroll
  for (int g = 0; g < 4; ++g) bv[g] = bias[g * 256 + u];

  v8f creg;
  if (init_zero) {
#pragma unroll
    for (int r = 0; r < 8; ++r) creg[r] = 0.0f;
    for (int i = tid; i < 32 * 256; i += 1024) h_sh[i] = (__bf16)0.0f;
  } else {
#pragma unroll
    for (int r = 0; r < 8; ++r) {
      const int b = m * 16 + r + 8 * half;
      creg[r] = hc_in[8192 + b * 256 + u];
    }
    for (int i = tid; i < 32 * 256; i += 1024) h_sh[i] = (__bf16)hc_in[i];
  }
  __syncthreads();

  for (int t = 0; t < L; ++t) {
    v8f acc[4];
#pragma unroll
    for (int g = 0; g < 4; ++g)
#pragma unroll
      for (int r = 0; r < 8; ++r) acc[g][r] = bv[g];

    for (int k = 0; k < 8; ++k) {  // K = 256 in 8 x 32 steps
      SCHED_FENCE();  // cap live operand registers to one k-step
      const __bf16* xrow = x_seq + (size_t)t * 8192 + (size_t)(m * 16 + lr) * 256 + k * 32;
      ABu ax;
      ax.h[0] = *(const v8bf*)(xrow + 16 * half);
      ax.h[1] = *(const v8bf*)(xrow + 16 * half + 8);
      const __bf16* hrow = h_sh + (size_t)(m * 16 + lr) * 256 + k * 32;
      ABu ah;
      ah.h[0] = *(const v8bf*)(hrow + 16 * half);
      ah.h[1] = *(const v8bf*)(hrow + 16 * half + 8);
#pragma unroll
      for (int g = 0; g < 4; ++g) {
        const int n = g * 256 + u;
        const v16bf bih = *(const v16bf*)(Wih + (size_t)n * 256 + k * 32 + 16 * half);
        acc[g] = WMMA_BF16(ax.v, bih, acc[g]);
        const v16bf bhh = *(const v16bf*)(Whh + (size_t)n * 256 + k * 32 + 16 * half);
        acc[g] = WMMA_BF16(ah.v, bhh, acc[g]);
      }
    }

    __syncthreads();  // all waves done reading h_sh for this step
#pragma unroll
    for (int r = 0; r < 8; ++r) {
      const float iv = sigmf(acc[0][r]);
      const float fv = sigmf(acc[1][r]);
      const float gv = tanhf(acc[2][r]);
      const float ov = sigmf(acc[3][r]);
      const float cn = fv * creg[r] + iv * gv;
      creg[r] = cn;
      const float hv = ov * tanhf(cn);
      const int b = m * 16 + r + 8 * half;
      const __bf16 hb = (__bf16)hv;
      h_sh[b * 256 + u] = hb;
      hs_out[(size_t)t * 8192 + b * 256 + u] = hb;
    }
    __syncthreads();  // h_sh visible to all waves for next step
  }

  // final (h, c) handoff
  for (int i = tid; i < 32 * 256; i += 1024) hc_out[i] = (float)h_sh[i];
#pragma unroll
  for (int r = 0; r < 8; ++r) {
    const int b = m * 16 + r + 8 * half;
    hc_out[8192 + b * 256 + u] = creg[r];
  }
}

// ---------------- zero out[:, 0, :] ----------------
__global__ __launch_bounds__(256) void zero_t0_kernel(float* __restrict__ out) {
  size_t tid = (size_t)blockIdx.x * 256 + threadIdx.x;
  if (tid >= (size_t)32 * 32000) return;
  size_t b = tid / 32000, v = tid % 32000;
  __builtin_nontemporal_store(0.0f, &out[b * (size_t)128 * 32000 + v]);
}

// ---------------- projection: out[b, t+1, :] = hs[t*32+b] @ fcW^T + fcb ----------------
// hs: [4064][256] bf16 (row = t*32+b), fcW: [32000][256] bf16, out: [32][128][32000] f32.
// Block = 8 waves; each wave computes a 32x64 strip (2 m-tiles x 4 n-tiles),
// K = 256 in 8 WMMA k-steps. fc_W tiles reused x2, hs tiles reused x4.
// Output is streamed with non-temporal stores (524 MB write-once >> 192 MB L2).
__global__ __launch_bounds__(256) void proj_kernel(const __bf16* __restrict__ hs,
                                                   const __bf16* __restrict__ fcW,
                                                   const float* __restrict__ fcb,
                                                   float* __restrict__ out) {
  const int w = threadIdx.x >> 5, lane = threadIdx.x & 31;
  const int half = lane >> 4, lr = lane & 15;
  const int mtile0 = blockIdx.y * 2;           // 254 m-tiles total (4064/16)
  const int nt0 = blockIdx.x * 32 + w * 4;     // n-tile base (2000 n-tiles total)

  v8f acc[2][4];
#pragma unroll
  for (int i = 0; i < 2; ++i)
#pragma unroll
    for (int j = 0; j < 4; ++j)
#pragma unroll
      for (int r = 0; r < 8; ++r) acc[i][j][r] = 0.0f;

  for (int k = 0; k < 8; ++k) {
    SCHED_FENCE();  // cap live operand registers to one k-step
    ABu a[2];
#pragma unroll
    for (int i = 0; i < 2; ++i) {
      const __bf16* arow = hs + (size_t)((mtile0 + i) * 16 + lr) * 256 + k * 32;
      a[i].h[0] = *(const v8bf*)(arow + 16 * half);
      a[i].h[1] = *(const v8bf*)(arow + 16 * half + 8);
    }
#pragma unroll
    for (int j = 0; j < 4; ++j) {
      const int nt = nt0 + j;          // wave-uniform -> EXEC stays all-ones
      if (nt < 2000) {
        const v16bf bmat =
            *(const v16bf*)(fcW + (size_t)(nt * 16 + lr) * 256 + k * 32 + 16 * half);
        acc[0][j] = WMMA_BF16(a[0].v, bmat, acc[0][j]);
        acc[1][j] = WMMA_BF16(a[1].v, bmat, acc[1][j]);
      }
    }
  }

#pragma unroll
  for (int j = 0; j < 4; ++j) {
    const int nt = nt0 + j;
    if (nt >= 2000) continue;
    const int n = nt * 16 + lr;
    const float bvv = fcb[n];
#pragma unroll
    for (int i = 0; i < 2; ++i)
#pragma unroll
      for (int r = 0; r < 8; ++r) {
        const int mrow = (mtile0 + i) * 16 + r + 8 * half;  // = t*32 + b
        const int b = mrow & 31, t = mrow >> 5;
        __builtin_nontemporal_store(
            acc[i][j][r] + bvv,
            &out[(size_t)b * 128 * 32000 + (size_t)(t + 1) * 32000 + n]);
      }
  }
}

// ---------------- host launch ----------------
extern "C" void kernel_launch(void* const* d_in, const int* in_sizes, int n_in,
                              void* d_out, int out_size, void* d_ws, size_t ws_size,
                              hipStream_t stream) {
  (void)in_sizes; (void)n_in; (void)out_size; (void)ws_size;
  const int*   src     = (const int*)d_in[0];
  const int*   tgt     = (const int*)d_in[1];
  const float* enc_emb = (const float*)d_in[2];
  const float* enc_Wih = (const float*)d_in[3];
  const float* enc_Whh = (const float*)d_in[4];
  const float* enc_b   = (const float*)d_in[5];
  const float* dec_emb = (const float*)d_in[6];
  const float* dec_Wih = (const float*)d_in[7];
  const float* dec_Whh = (const float*)d_in[8];
  const float* dec_b   = (const float*)d_in[9];
  const float* fc_W    = (const float*)d_in[10];
  const float* fc_b    = (const float*)d_in[11];
  float* out = (float*)d_out;

  char* ws = (char*)d_ws;
  __bf16* x_enc  = (__bf16*)(ws + 0);                  // 128*32*256*2 = 2 MB
  __bf16* x_dec  = (__bf16*)(ws + (2u << 20));         // 2 MB
  __bf16* hs_dec = (__bf16*)(ws + (4u << 20));         // 2 MB
  __bf16* hs_enc = (__bf16*)(ws + (6u << 20));         // 2 MB (scratch sink)
  __bf16* wih_e  = (__bf16*)(ws + (8u << 20));         // 512 KB each
  __bf16* whh_e  = (__bf16*)(ws + (8u << 20) + 1 * 524288u);
  __bf16* wih_d  = (__bf16*)(ws + (8u << 20) + 2 * 524288u);
  __bf16* whh_d  = (__bf16*)(ws + (8u << 20) + 3 * 524288u);
  __bf16* fcw_bf = (__bf16*)(ws + (10u << 20));        // 32000*256*2 = ~15.6 MB
  float*  hc_e   = (float*)(ws + (27u << 20));         // 64 KB  (h then c)
  float*  hc_d   = (float*)(ws + (27u << 20) + 65536u);

  // 1) weight conversions fp32 -> bf16
  const size_t wsz = 1024 * 256;
  cvt_bf16_kernel<<<(wsz + 255) / 256, 256, 0, stream>>>(enc_Wih, wih_e, wsz);
  cvt_bf16_kernel<<<(wsz + 255) / 256, 256, 0, stream>>>(enc_Whh, whh_e, wsz);
  cvt_bf16_kernel<<<(wsz + 255) / 256, 256, 0, stream>>>(dec_Wih, wih_d, wsz);
  cvt_bf16_kernel<<<(wsz + 255) / 256, 256, 0, stream>>>(dec_Whh, whh_d, wsz);
  const size_t fsz = (size_t)32000 * 256;
  cvt_bf16_kernel<<<(fsz + 255) / 256, 256, 0, stream>>>(fc_W, fcw_bf, fsz);

  // 2) embedding gathers
  {
    size_t ne = (size_t)128 * 32 * 256;
    embed_kernel<<<(ne + 255) / 256, 256, 0, stream>>>(src, 128, enc_emb, x_enc, 128);
    size_t nd = (size_t)127 * 32 * 256;
    embed_kernel<<<(nd + 255) / 256, 256, 0, stream>>>(tgt, 128, dec_emb, x_dec, 127);
  }

  // 3) encoder LSTM (single persistent 32-wave workgroup), decoder chained via hc_e
  lstm_kernel<<<1, 1024, 0, stream>>>(x_enc, wih_e, whh_e, enc_b, hc_e, 1, hc_e, hs_enc, 128);
  lstm_kernel<<<1, 1024, 0, stream>>>(x_dec, wih_d, whh_d, dec_b, hc_e, 0, hc_d, hs_dec, 127);

  // 4) outputs[:, 0, :] = 0
  zero_t0_kernel<<<((size_t)32 * 32000 + 255) / 256, 256, 0, stream>>>(out);

  // 5) big projection GEMM with bias
  dim3 grid(63, 127);
  proj_kernel<<<grid, 256, 0, stream>>>(hs_dec, fcw_bf, fc_b, out);
}
